// DeltaNet_31877247271472
// MI455X (gfx1250) — matrix operand order repaired and checked by reading, not compiled, not run on hardware
//
#include <hip/hip_runtime.h>

// ---------------- problem constants ----------------
#define B_   2
#define L_   4096
#define D_   1024
#define H_   8
#define DK_  128
#define DV_  128
#define CHUNK_ 32
#define NCH_ (L_/CHUNK_)       // 128 chunks
#define DVSPLIT_ 4
#define DVSEG_  (DV_/DVSPLIT_) // 32 dv columns per block

typedef __attribute__((ext_vector_type(16))) _Float16 v16h;
typedef __attribute__((ext_vector_type(8)))  _Float16 v8h;
typedef __attribute__((ext_vector_type(4)))  _Float16 v4h;
typedef __attribute__((ext_vector_type(8)))  float    v8f;
typedef __attribute__((ext_vector_type(4)))  float    v4f;

// gfx1250 async global->LDS path (guarded; falls back to sync copies)
#if defined(__AMDGCN__) && __has_builtin(__builtin_amdgcn_global_load_async_to_lds_b128) && __has_builtin(__builtin_amdgcn_s_wait_asynccnt)
#define ASYNC_LDS 1
// exact parameter types per hipcc diagnostic: vector_size(16) int, AS1 / AS3
typedef int v4i_ __attribute__((vector_size(16)));
typedef __attribute__((address_space(1))) v4i_* async_gptr;
typedef __attribute__((address_space(3))) v4i_* async_lptr;
#define ASYNC_B128(gsrc, ldst) \
  __builtin_amdgcn_global_load_async_to_lds_b128( \
      (async_gptr)(void*)(gsrc), (async_lptr)(void*)(ldst), 0, 0)
#else
#define ASYNC_LDS 0
#endif

// ---------------- WMMA wrapper (CDNA5, wave32) ----------------
__device__ __forceinline__ v8f wmma_f16(v16h a, v16h b, v8f c) {
  return __builtin_amdgcn_wmma_f32_16x16x32_f16(false, a, false, b, (short)0, c,
                                                false, false);
}

// ----- fragment loaders (wave32; A/B use contiguous k = 16*(lane/16) + s) -----
// A from row-major [m][k]: two contiguous 16B LDS loads per lane.
__device__ __forceinline__ v16h frag_a_row(const _Float16* p, int ld, int m0, int k0) {
  int lane = threadIdx.x & 31;
  int m = m0 + (lane & 15);
  int kb = k0 + ((lane >> 4) << 4);
  const _Float16* q = p + m * ld + kb;
  union { v16h v; v8h h[2]; } u;
  u.h[0] = *(const v8h*)q;          // halves 0..7  -> K = kb..kb+7
  u.h[1] = *(const v8h*)(q + 8);    // halves 8..15 -> K = kb+8..kb+15
  return u.v;
}
// A from k-major source [k][m] (strided; used sparingly)
__device__ __forceinline__ v16h frag_a_kmajor(const _Float16* p, int ld, int m0, int k0) {
  int lane = threadIdx.x & 31;
  int m = m0 + (lane & 15);
  int kb = k0 + ((lane >> 4) << 4);
  v16h a;
#pragma unroll
  for (int i = 0; i < 16; ++i) {
    int kk = kb + i;
    a[i] = p[kk * ld + m];
  }
  return a;
}
// B from N-major [n][k]: 16 contiguous halves per lane (two 16B loads).
__device__ __forceinline__ v16h frag_b_nmajor(const _Float16* p, int ld, int k0, int n0) {
  int lane = threadIdx.x & 31;
  int n = n0 + (lane & 15);
  int kb = k0 + ((lane >> 4) << 4);
  const _Float16* q = p + n * ld + kb;
  union { v16h v; v8h h[2]; } u;
  u.h[0] = *(const v8h*)q;
  u.h[1] = *(const v8h*)(q + 8);
  return u.v;
}
// B from N-major f32 source (state S stored [dv][dk]): vector loads + cvt.
__device__ __forceinline__ v16h frag_b_nmajor_f32(const float* p, int ld, int k0, int n0) {
  int lane = threadIdx.x & 31;
  int n = n0 + (lane & 15);
  int kb = k0 + ((lane >> 4) << 4);
  const float* q = p + n * ld + kb;
  v4f c0 = *(const v4f*)q;
  v4f c1 = *(const v4f*)(q + 4);
  v4f c2 = *(const v4f*)(q + 8);
  v4f c3 = *(const v4f*)(q + 12);
  v16h b;
#pragma unroll
  for (int i = 0; i < 4; ++i) {
    b[i]      = (_Float16)c0[i];
    b[4 + i]  = (_Float16)c1[i];
    b[8 + i]  = (_Float16)c2[i];
    b[12 + i] = (_Float16)c3[i];
  }
  return b;
}
// C/D on N-major f32 [n][m]: 8 contiguous floats per lane.
__device__ __forceinline__ v8f frag_c_nmajor(const float* p, int ld, int m0, int n0) {
  int lane = threadIdx.x & 31;
  int n = n0 + (lane & 15);
  int mb = m0 + ((lane >> 4) << 3);
  const float* q = p + n * ld + mb;
  union { v8f v; v4f h[2]; } u;
  u.h[0] = *(const v4f*)q;
  u.h[1] = *(const v4f*)(q + 4);
  return u.v;
}
__device__ __forceinline__ void store_d_nmajor(float* p, int ld, int m0, int n0, v8f d) {
  int lane = threadIdx.x & 31;
  int n = n0 + (lane & 15);
  int mb = m0 + ((lane >> 4) << 3);
  float* q = p + n * ld + mb;
  union { v8f v; v4f h[2]; } u;
  u.v = d;
  *(v4f*)q = u.h[0];
  *(v4f*)(q + 4) = u.h[1];
}

__device__ __forceinline__ float sigf(float x) { return 1.0f / (1.0f + __expf(-x)); }

// ---------------- f32 -> f16 conversion ----------------
__global__ void f32_to_f16_kernel(const float* __restrict__ x,
                                  _Float16* __restrict__ y, int n) {
  int i = blockIdx.x * blockDim.x + threadIdx.x;
  if (i < n) y[i] = (_Float16)x[i];
}

// ---------------- WMMA GEMM: C[MxN] = A[MxK] * B[KxN] ----------------
// 128x128 block tile, K-slab 64, 8 waves. B staged N-major in LDS so all
// fragment reads are ds_load_b128. A tile staged via async global->LDS when
// available.
template <typename OutT>
__global__ void gemm_f16_kernel(const _Float16* __restrict__ A,
                                const _Float16* __restrict__ Bm,
                                OutT* __restrict__ C, int M, int N, int K) {
  __shared__ _Float16 sA[128 * 64];    // [m][k]
  __shared__ _Float16 sBt[128 * 64];   // [n][k]  (N-major)
  int tid = threadIdx.x;
  int lane = tid & 31, wave = tid >> 5;
  int bm = blockIdx.x * 128, bn = blockIdx.y * 128;

  v8f acc[8];
#pragma unroll
  for (int i = 0; i < 8; ++i) acc[i] = (v8f){};

  for (int k0 = 0; k0 < K; k0 += 64) {
    __syncthreads();
    {  // stage A tile: 128x64 halves, 32 halves/thread
      int f = tid * 32;
      int r = f >> 6, c = f & 63;
      const _Float16* g = A + (size_t)(bm + r) * K + k0 + c;
#if ASYNC_LDS
#pragma unroll
      for (int u2 = 0; u2 < 4; ++u2)
        ASYNC_B128(g + u2 * 8, &sA[f + u2 * 8]);
#else
#pragma unroll
      for (int u2 = 0; u2 < 4; ++u2)
        *(uint4*)&sA[f + u2 * 8] = *(const uint4*)(g + u2 * 8);
#endif
    }
    {  // stage B tile transposed: global rows (k, n..n+31) -> sBt[n][k]
      int f = tid * 32;
      int r = f >> 7, c = f & 127;          // r: k row 0..63, c: n base
      const _Float16* g = Bm + (size_t)(k0 + r) * N + bn + c;
      union { uint4 u[4]; _Float16 h[32]; } tmp;
#pragma unroll
      for (int u2 = 0; u2 < 4; ++u2) tmp.u[u2] = *(const uint4*)(g + u2 * 8);
#pragma unroll
      for (int i = 0; i < 32; ++i) sBt[(c + i) * 64 + r] = tmp.h[i];
    }
    if (k0 + 64 < K) {  // gfx1250 global_prefetch_b8 of next K slab
      __builtin_prefetch(A + (size_t)(bm + (tid >> 1)) * K + (k0 + 64), 0, 1);
      __builtin_prefetch(Bm + (size_t)(k0 + 64 + (tid >> 2)) * N + bn + ((tid & 3) << 5), 0, 1);
    }
#if ASYNC_LDS
    __builtin_amdgcn_s_wait_asynccnt(0);
#endif
    __syncthreads();
#pragma unroll
    for (int ks = 0; ks < 64; ks += 32) {
      v16h af = frag_a_row(sA, 64, wave * 16, ks);
#pragma unroll
      for (int nt = 0; nt < 8; ++nt) {
        v16h bf = frag_b_nmajor(sBt, 64, ks, nt * 16);
        acc[nt] = wmma_f16(af, bf, acc[nt]);
      }
    }
  }
  int mb = bm + wave * 16 + ((lane >> 4) << 3);
#pragma unroll
  for (int nt = 0; nt < 8; ++nt) {
    int n = bn + nt * 16 + (lane & 15);
#pragma unroll
    for (int r = 0; r < 8; ++r) C[(size_t)(mb + r) * N + n] = (OutT)acc[nt][r];
  }
}

// ---------------- causal depthwise conv (K=4) + SiLU ----------------
__global__ void conv_silu_kernel(const _Float16* __restrict__ x,
                                 const float* __restrict__ w,
                                 _Float16* __restrict__ y) {
  int idx = blockIdx.x * 256 + threadIdx.x;      // over B*L*D
  int ch = idx & (D_ - 1);
  int t  = (idx >> 10) & (L_ - 1);
  int b  = idx >> 22;
  float acc = 0.f;
#pragma unroll
  for (int i = 0; i < 4; ++i) {
    int tt = t - 3 + i;
    float xv = (tt >= 0) ? (float)x[((size_t)b * L_ + tt) * D_ + ch] : 0.f;
    acc += xv * w[ch * 4 + i];
  }
  y[idx] = (_Float16)(acc * sigf(acc));
}

// ---------------- beta = sigmoid(hs@b_w); g = sigmoid(hs@mix_w + biases) ----
__global__ void beta_g_kernel(const float* __restrict__ hs,
                              const float* __restrict__ b_w,
                              const float* __restrict__ mix_w,
                              const float* __restrict__ mix_b,
                              const float* __restrict__ mix_bias,
                              float* __restrict__ beta, float* __restrict__ g) {
  int idx = blockIdx.x * 256 + threadIdx.x;  // over (B*L)*H
  int h = idx & (H_ - 1);
  int tok = idx >> 3;
  const float* hp = hs + (size_t)tok * D_;
  float sb = 0.f, sg = 0.f;
  for (int d = 0; d < D_; ++d) {
    float x = hp[d];
    sb += x * b_w[d * H_ + h];
    sg += x * mix_w[d * H_ + h];
  }
  beta[idx] = sigf(sb);
  g[idx]    = sigf(sg + mix_b[h] + mix_bias[h]);
}

// ---------------- chunkwise delta rule ----------------
// grid: B_*H_*DVSPLIT_ blocks, 256 threads (8 waves). Each block owns a
// 32-wide dv slice of the state S ([dv][dk] f32, LDS-resident) and scans 128
// chunks sequentially. Operand layouts are chosen so that WMMA A fragments
// read row-major tiles and B fragments read N-major tiles -> ds_load_b128.
__global__ void delta_kernel(const _Float16* __restrict__ qc,
                             const _Float16* __restrict__ kc,
                             const _Float16* __restrict__ vc,
                             const float* __restrict__ beta,
                             float* __restrict__ o_core) {
  __shared__ float    St[DVSEG_ * 128];   // state, [dv][dk]
  __shared__ float    attnL[32 * 32];     // triangular resolve workspace
  __shared__ _Float16 qL[32 * 128];       // [token][dk]
  __shared__ _Float16 kL[32 * 128];       // [token][dk]  (N-major view of k^T)
  __shared__ _Float16 kbT[128 * 32];      // [dk][token]  (N-major kb)
  __shared__ _Float16 wnL[32 * 128];      // [token][dk]  -(attn@kb)
  __shared__ _Float16 vT[DVSEG_ * 32];    // [dv][token]  v*beta
  __shared__ _Float16 uT[DVSEG_ * 32];    // [dv][token]
  __shared__ _Float16 aH[32 * 32];        // inverted attn (f16, row-major)
  __shared__ _Float16 a2H[32 * 32];       // tril(q k^T)   (f16, row-major)
  __shared__ float    red[512];

  int tid = threadIdx.x, lane = tid & 31, wave = tid >> 5;
  int blk = blockIdx.x;
  int gseg = blk & (DVSPLIT_ - 1);
  int h = (blk >> 2) & (H_ - 1);
  int b = blk >> 5;
  int dv0 = gseg * DVSEG_;
  const size_t headoff = (size_t)h * DK_;

  for (int i = tid; i < DVSEG_ * 128; i += 256) St[i] = 0.f;
  __syncthreads();

  v8f zero = (v8f){};

  for (int c = 0; c < NCH_; ++c) {
    int t0 = c * CHUNK_;

    // ---- P1: load q/k/v, l2-normalize q,k; kb=k*beta (kbT); v*=beta (vT) --
    {
      int row = tid >> 3, seg = tid & 7;   // 32 rows x 8 segs x 16 cols
      size_t gbase = ((size_t)b * L_ + t0 + row) * D_ + headoff;
      _Float16 qv[16], kv[16];
      float sq = 0.f, sk = 0.f;
#pragma unroll
      for (int i = 0; i < 16; ++i) {
        int d = seg * 16 + i;
        _Float16 qh = qc[gbase + d], kh = kc[gbase + d];
        qv[i] = qh; kv[i] = kh;
        float qf = (float)qh, kf = (float)kh;
        sq += qf * qf; sk += kf * kf;
      }
      red[row * 8 + seg] = sq;
      red[256 + row * 8 + seg] = sk;
      __syncthreads();
      float ssq = 1e-12f, ssk = 1e-12f;
#pragma unroll
      for (int s2 = 0; s2 < 8; ++s2) {
        ssq += red[row * 8 + s2];
        ssk += red[256 + row * 8 + s2];
      }
      float rq = rsqrtf(ssq), rk = rsqrtf(ssk);
      float bv = beta[((size_t)b * L_ + t0 + row) * H_ + h];
#pragma unroll
      for (int i = 0; i < 16; ++i) {
        int d = seg * 16 + i;
        float kn = (float)kv[i] * rk;
        qL[row * 128 + d]  = (_Float16)((float)qv[i] * rq);
        kL[row * 128 + d]  = (_Float16)kn;
        kbT[d * 32 + row]  = (_Float16)(kn * bv);
      }
      if (seg < 2) {
#pragma unroll
        for (int i = 0; i < 16; ++i) {
          int d = seg * 16 + i;
          vT[d * 32 + row] = (_Float16)((float)vc[gbase + dv0 + d] * bv);
        }
      }
    }
    __syncthreads();

    // ---- P2: attn = strict_tril(-(kb @ k^T))  (waves 0..3, uniform) ----
    if (wave < 4) {
      int m0 = (wave >> 1) * 16, n0 = (wave & 1) * 16;
      v8f acc = zero;
#pragma unroll
      for (int ks = 0; ks < 4; ++ks) {
        v16h af = frag_a_kmajor(kbT, 32, m0, ks * 32);     // kb[m=token][k=dk]
        v16h bf = frag_b_nmajor(kL, 128, ks * 32, n0);     // k^T as B (N-major)
        acc = wmma_f16(af, bf, acc);
      }
      int n = n0 + (lane & 15), mb = m0 + ((lane >> 4) << 3);
#pragma unroll
      for (int r = 0; r < 8; ++r) {
        int m = mb + r;
        attnL[m * 32 + n] = (n < m) ? -acc[r] : 0.f;
      }
    }
    __syncthreads();

    // ---- P3: sequential unit-triangular resolve (wave 0) + identity ----
    if (wave == 0) {
      int k = lane;
      for (int i = 1; i < 32; ++i) {
        float s = 0.f;
        for (int j = 0; j < i; ++j) s += attnL[i * 32 + j] * attnL[j * 32 + k];
        __threadfence_block();
        if (k < i) attnL[i * 32 + k] += s;
        __threadfence_block();
      }
      attnL[k * 32 + k] += 1.0f;
    }
    __syncthreads();
    {  // attnL -> aH (f16), 4 contiguous elems/thread
      int i0 = tid * 4;
      v4f a4 = *(const v4f*)&attnL[i0];
      v4h h4;
#pragma unroll
      for (int j = 0; j < 4; ++j) h4[j] = (_Float16)a4[j];
      *(v4h*)&aH[i0] = h4;
    }
    __syncthreads();

    // ---- P4: wn = -(attn @ kb)  (32 x 128, 16 tiles, all 8 waves) ----
    {
      int tt = wave * 2;
#pragma unroll
      for (int u2 = 0; u2 < 2; ++u2, ++tt) {
        int mi = tt >> 3, ni = tt & 7;
        v16h af = frag_a_row(aH, 32, mi * 16, 0);
        v16h bf = frag_b_nmajor(kbT, 32, 0, ni * 16);      // kb as B (N-major)
        v8f acc = wmma_f16(af, bf, zero);
        int n = ni * 16 + (lane & 15), mb = mi * 16 + ((lane >> 4) << 3);
#pragma unroll
        for (int r = 0; r < 8; ++r)
          wnL[(mb + r) * 128 + n] = (_Float16)(-acc[r]);
      }
    }
    __syncthreads();

    // ---- P5: u = attn@v + wn@S (waves 0-3) || P6: attn2 (waves 4-7) ----
    if (wave < 4) {
      int mi = wave >> 1, ni = wave & 1;
      v16h af = frag_a_row(aH, 32, mi * 16, 0);
      v16h bf = frag_b_nmajor(vT, 32, 0, ni * 16);
      v8f acc = wmma_f16(af, bf, zero);
#pragma unroll
      for (int ks = 0; ks < 4; ++ks) {
        v16h a2 = frag_a_row(wnL, 128, mi * 16, ks * 32);
        v16h b2 = frag_b_nmajor_f32(St, 128, ks * 32, ni * 16);
        acc = wmma_f16(a2, b2, acc);
      }
      // store u N-major: uT[n][m], 8 contiguous halves per lane
      int n = ni * 16 + (lane & 15), mb = mi * 16 + ((lane >> 4) << 3);
      v8h pk;
#pragma unroll
      for (int r = 0; r < 8; ++r) pk[r] = (_Float16)acc[r];
      *(v8h*)&uT[n * 32 + mb] = pk;
    } else {
      int w2 = wave - 4;
      int m0 = (w2 >> 1) * 16, n0 = (w2 & 1) * 16;
      v8f acc = zero;
#pragma unroll
      for (int ks = 0; ks < 4; ++ks) {
        v16h af = frag_a_row(qL, 128, m0, ks * 32);
        v16h bf = frag_b_nmajor(kL, 128, ks * 32, n0);
        acc = wmma_f16(af, bf, acc);
      }
      int n = n0 + (lane & 15), mb = m0 + ((lane >> 4) << 3);
#pragma unroll
      for (int r = 0; r < 8; ++r) {
        int m = mb + r;
        a2H[m * 32 + n] = (n <= m) ? (_Float16)acc[r] : (_Float16)0.f;
      }
    }
    __syncthreads();

    // ---- P7: o = q@S + attn2@u  (waves 0-3) ----
    if (wave < 4) {
      int mi = wave >> 1, ni = wave & 1;
      v8f acc = zero;
#pragma unroll
      for (int ks = 0; ks < 4; ++ks) {
        v16h af = frag_a_row(qL, 128, mi * 16, ks * 32);
        v16h bf = frag_b_nmajor_f32(St, 128, ks * 32, ni * 16);
        acc = wmma_f16(af, bf, acc);
      }
      v16h af2 = frag_a_row(a2H, 32, mi * 16, 0);
      v16h bf2 = frag_b_nmajor(uT, 32, 0, ni * 16);
      acc = wmma_f16(af2, bf2, acc);
      int n = ni * 16 + (lane & 15), mb = mi * 16 + ((lane >> 4) << 3);
#pragma unroll
      for (int r = 0; r < 8; ++r) {
        int m = mb + r;
        o_core[((size_t)b * L_ + t0 + m) * D_ + h * DV_ + dv0 + n] = acc[r];
      }
    }
    __syncthreads();

    // ---- P8: S += k^T @ u  (8x2 tiles, one dk row-strip per wave) ----
    {
      int mi = wave;
      v16h af = frag_a_kmajor(kL, 128, mi * 16, 0);        // k^T[m=dk][kk=token]
#pragma unroll
      for (int ni = 0; ni < 2; ++ni) {
        v8f cfr = frag_c_nmajor(St, 128, mi * 16, ni * 16);
        v16h bf = frag_b_nmajor(uT, 32, 0, ni * 16);
        v8f d = wmma_f16(af, bf, cfr);
        store_d_nmajor(St, 128, mi * 16, ni * 16, d);
      }
    }
    __syncthreads();
  }
}

// ---------------- gate mix + RMS norm -> f16 for final GEMM ----------------
__global__ void mix_norm_kernel(const float* __restrict__ o_core,
                                const _Float16* __restrict__ vc,
                                const float* __restrict__ g,
                                const float* __restrict__ o_norm_w,
                                _Float16* __restrict__ out) {
  __shared__ float red[128];
  int bh = blockIdx.x;                    // (b*L + t)*H + h
  int dv = threadIdx.x;
  size_t base = (size_t)(bh >> 3) * D_ + (size_t)(bh & (H_ - 1)) * DV_;
  float gv = g[bh];
  float o = gv * o_core[base + dv] + (1.f - gv) * (float)vc[base + dv];
  red[dv] = o * o;
  __syncthreads();
  for (int s = 64; s > 0; s >>= 1) {
    if (dv < s) red[dv] += red[dv + s];
    __syncthreads();
  }
  float mean = red[0] * (1.0f / 128.0f);
  out[base + dv] = (_Float16)(o * rsqrtf(mean + 1e-5f) * o_norm_w[dv]);
}

// ---------------- host-side orchestration ----------------
extern "C" void kernel_launch(void* const* d_in, const int* in_sizes, int n_in,
                              void* d_out, int out_size, void* d_ws, size_t ws_size,
                              hipStream_t stream) {
  const float* hs       = (const float*)d_in[0];
  const float* q_w      = (const float*)d_in[1];
  const float* k_w      = (const float*)d_in[2];
  const float* v_w      = (const float*)d_in[3];
  const float* conv_q_w = (const float*)d_in[4];
  const float* conv_k_w = (const float*)d_in[5];
  const float* conv_v_w = (const float*)d_in[6];
  const float* b_w      = (const float*)d_in[7];
  const float* mix_w    = (const float*)d_in[8];
  const float* mix_b    = (const float*)d_in[9];
  const float* mix_bias = (const float*)d_in[10];
  const float* o_norm_w = (const float*)d_in[11];
  const float* o_w      = (const float*)d_in[12];
  float* out = (float*)d_out;

  const size_t NTOK = (size_t)B_ * L_;          // 8192
  const size_t NELT = NTOK * D_;                // 8,388,608

  char* wsp = (char*)d_ws;
  auto alloc = [&](size_t bytes) -> char* {
    char* p = wsp;
    wsp += (bytes + 255) & ~(size_t)255;
    return p;
  };
  _Float16* hs16   = (_Float16*)alloc(NELT * 2);
  _Float16* wq16   = (_Float16*)alloc((size_t)D_ * D_ * 2);
  _Float16* wk16   = (_Float16*)alloc((size_t)D_ * D_ * 2);
  _Float16* wv16   = (_Float16*)alloc((size_t)D_ * D_ * 2);
  _Float16* wo16   = (_Float16*)alloc((size_t)D_ * D_ * 2);
  _Float16* qlin   = (_Float16*)alloc(NELT * 2);
  _Float16* klin   = (_Float16*)alloc(NELT * 2);
  _Float16* vlin   = (_Float16*)alloc(NELT * 2);
  _Float16* qc16   = (_Float16*)alloc(NELT * 2);
  _Float16* kc16   = (_Float16*)alloc(NELT * 2);
  _Float16* vc16   = (_Float16*)alloc(NELT * 2);
  float*    betab  = (float*)alloc(NTOK * H_ * 4);
  float*    gbuf   = (float*)alloc(NTOK * H_ * 4);
  float*    o_core = (float*)alloc(NELT * 4);
  _Float16* omix16 = (_Float16*)alloc(NELT * 2);

  // 1) convert activations & weights to f16
  f32_to_f16_kernel<<<(int)(NELT / 256), 256, 0, stream>>>(hs, hs16, (int)NELT);
  f32_to_f16_kernel<<<D_ * D_ / 256, 256, 0, stream>>>(q_w, wq16, D_ * D_);
  f32_to_f16_kernel<<<D_ * D_ / 256, 256, 0, stream>>>(k_w, wk16, D_ * D_);
  f32_to_f16_kernel<<<D_ * D_ / 256, 256, 0, stream>>>(v_w, wv16, D_ * D_);
  f32_to_f16_kernel<<<D_ * D_ / 256, 256, 0, stream>>>(o_w, wo16, D_ * D_);

  // 2) q/k/v projections (WMMA GEMMs), f16 out
  dim3 ggrid((unsigned)(NTOK / 128), D_ / 128);
  gemm_f16_kernel<_Float16><<<ggrid, 256, 0, stream>>>(hs16, wq16, qlin, (int)NTOK, D_, D_);
  gemm_f16_kernel<_Float16><<<ggrid, 256, 0, stream>>>(hs16, wk16, klin, (int)NTOK, D_, D_);
  gemm_f16_kernel<_Float16><<<ggrid, 256, 0, stream>>>(hs16, wv16, vlin, (int)NTOK, D_, D_);

  // 3) causal depthwise conv + SiLU
  conv_silu_kernel<<<(int)(NELT / 256), 256, 0, stream>>>(qlin, conv_q_w, qc16);
  conv_silu_kernel<<<(int)(NELT / 256), 256, 0, stream>>>(klin, conv_k_w, kc16);
  conv_silu_kernel<<<(int)(NELT / 256), 256, 0, stream>>>(vlin, conv_v_w, vc16);

  // 4) beta and gate projections
  beta_g_kernel<<<(int)(NTOK * H_ / 256), 256, 0, stream>>>(hs, b_w, mix_w, mix_b,
                                                            mix_bias, betab, gbuf);

  // 5) chunkwise delta rule (WMMA-resident state scan)
  delta_kernel<<<B_ * H_ * DVSPLIT_, 256, 0, stream>>>(qc16, kc16, vc16, betab, o_core);

  // 6) gate mix + RMS norm
  mix_norm_kernel<<<(int)(NTOK * H_), 128, 0, stream>>>(o_core, vc16, gbuf,
                                                        o_norm_w, omix16);

  // 7) output projection (WMMA GEMM), f32 out
  gemm_f16_kernel<float><<<ggrid, 256, 0, stream>>>(omix16, wo16, out, (int)NTOK, D_, D_);
}
